// TSingleHeadCausalSelfAttention_82429012345483
// MI455X (gfx1250) — matrix operand order repaired and checked by reading, not compiled, not run on hardware
//
#include <hip/hip_runtime.h>

// ---------------------------------------------------------------------------
// Single-head causal self-attention, T=4096, C=1024, MI455X (gfx1250).
// bf16 WMMA (v_wmma_f32_16x16x32_bf16) for all three matmuls, fp32 accum,
// intermediates in d_ws (~88MB, fits in 192MB L2). Compute-bound (~95 GFLOP
// vs ~200MB traffic ≈ 9us at 23.3 TB/s), so everything goes to matrix cores.
// ---------------------------------------------------------------------------

#define T_DIM 4096
#define C_DIM 1024
#define C3    3072
#define NEG_INF_F (-1.0e30f)

typedef __bf16          bf16_t;
typedef bf16_t          v16bf __attribute__((ext_vector_type(16)));
typedef unsigned short  v16us __attribute__((ext_vector_type(16)));
typedef float           v8f   __attribute__((ext_vector_type(8)));
typedef int             gv4i  __attribute__((vector_size(16)));   // GCC-style int4

#if __has_builtin(__builtin_amdgcn_global_load_async_to_lds_b128) && \
    __has_builtin(__builtin_amdgcn_s_wait_asynccnt)
#define HAVE_ASYNC 1
#else
#define HAVE_ASYNC 0
#endif

// float -> bf16 bits (RNE)
static __device__ __forceinline__ unsigned short f2bf(float f) {
    unsigned int u = __float_as_uint(f);
    u += 0x7FFFu + ((u >> 16) & 1u);
    return (unsigned short)(u >> 16);
}
static __device__ __forceinline__ float bf2f(unsigned short h) {
    return __uint_as_float(((unsigned int)h) << 16);
}

static __device__ __forceinline__ v8f wmma_bf16(v16us a, v16us b, v8f c) {
    return __builtin_amdgcn_wmma_f32_16x16x32_bf16(
        false, __builtin_bit_cast(v16bf, a),
        false, __builtin_bit_cast(v16bf, b),
        (short)0, c, false, false);
}

// Load a 16-element bf16 fragment as two aligned 16B LDS chunks -> 2x ds_load_b128
static __device__ __forceinline__ v16us ld2(const unsigned short* p0,
                                            const unsigned short* p1) {
    struct U { uint4 a, b; } u;
    u.a = *reinterpret_cast<const uint4*>(p0);
    u.b = *reinterpret_cast<const uint4*>(p1);
    return __builtin_bit_cast(v16us, u);
}

// Global(bf16) -> LDS 16B copy; async-to-LDS on toolchains that expose it.
// Builtin signature (from hipcc diagnostic): param0 = int4 AS(1)* (non-const).
static __device__ __forceinline__ void cp16_g2l(const unsigned short* g,
                                                unsigned short* l) {
#if HAVE_ASYNC
    __builtin_amdgcn_global_load_async_to_lds_b128(
        (__attribute__((address_space(1))) gv4i*)g,
        (__attribute__((address_space(3))) gv4i*)l, 0, 0);
#else
    *reinterpret_cast<uint4*>(l) = *reinterpret_cast<const uint4*>(g);
#endif
}
static __device__ __forceinline__ void wait_async() {
#if HAVE_ASYNC
    __builtin_amdgcn_s_wait_asynccnt(0);
#endif
}

// Tile config: 128x128 block tile, K-step 64 (2 WMMA K-chunks per stage),
// 256 threads = 8 waves; wave (wm, wn) owns 32x64 subtile = 2x4 f32 accs.
#define BM 128
#define BN 128
#define BK 64
#define LDST (BK + 8)     // 72 ushorts = 144B row stride (16B multiple)

// ISA fragment layouts (7.12.2), lane = tid&31, h = lane>>4, e = lane&15:
//  A (16x32 bf16): lane row = e; 16B chunks at [kc + h*8], [kc + 16 + h*8]
//  B (32x16 bf16): lane col = e; with n-major LDS tile Bt[n][k]:
//                  16B chunks at [kc + h*8], [kc + 16 + h*8]
//  C/D: VGPR v -> row = h*8 + v, col = e

// ===========================================================================
// Kernel 1: qkv = x @ W + b   (M=4096, N=3072, K=1024), fp32 in, bf16 out
// ===========================================================================
__global__ __launch_bounds__(256)
void qkv_gemm_kernel(const float* __restrict__ x, const float* __restrict__ W,
                     const float* __restrict__ bias, unsigned short* __restrict__ qkv) {
    __shared__ __align__(16) unsigned short As[BM][LDST];   // [m][k]
    __shared__ __align__(16) unsigned short Bt[BN][LDST];   // [n][k] (transposed)

    const int tid  = threadIdx.x;
    const int lane = tid & 31;
    const int h    = lane >> 4;
    const int e    = lane & 15;
    const int wid  = tid >> 5;
    const int wm   = wid & 3;
    const int wn   = wid >> 2;
    const int m0   = blockIdx.y * BM;
    const int n0   = blockIdx.x * BN;

    v8f acc[2][4] = {};

    for (int k0 = 0; k0 < C_DIM; k0 += BK) {
        // A tile: 128 rows x 64 k, fp32->bf16. 2048 float4, 8/thread.
        #pragma unroll
        for (int i = 0; i < 8; ++i) {
            int p   = tid + i * 256;
            int row = p >> 4;            // 16 float4 per row
            int c4  = (p & 15) << 2;
            const float* src = x + (size_t)(m0 + row) * C_DIM + k0 + c4;
            if (i == 0 && k0 + BK < C_DIM) __builtin_prefetch(src + BK, 0, 1);
            float4 f = *reinterpret_cast<const float4*>(src);
            unsigned short* d = &As[row][c4];
            d[0] = f2bf(f.x); d[1] = f2bf(f.y); d[2] = f2bf(f.z); d[3] = f2bf(f.w);
        }
        // B tile: 64 k-rows x 128 n, fp32->bf16, stored transposed. 8/thread.
        #pragma unroll
        for (int i = 0; i < 8; ++i) {
            int p   = tid + i * 256;
            int kr  = p >> 5;            // 32 float4 per k-row
            int c4  = (p & 31) << 2;
            const float* src = W + (size_t)(k0 + kr) * C3 + n0 + c4;
            if (i == 0 && k0 + BK < C_DIM) __builtin_prefetch(src + (size_t)BK * C3, 0, 1);
            float4 f = *reinterpret_cast<const float4*>(src);
            Bt[c4 + 0][kr] = f2bf(f.x);
            Bt[c4 + 1][kr] = f2bf(f.y);
            Bt[c4 + 2][kr] = f2bf(f.z);
            Bt[c4 + 3][kr] = f2bf(f.w);
        }
        __syncthreads();

        #pragma unroll
        for (int kc = 0; kc < BK; kc += 32) {
            v16us af[2], bf[4];
            #pragma unroll
            for (int i = 0; i < 2; ++i) {
                int row = wm * 32 + i * 16 + e;
                af[i] = ld2(&As[row][kc + h * 8], &As[row][kc + 16 + h * 8]);
            }
            #pragma unroll
            for (int j = 0; j < 4; ++j) {
                int col = wn * 64 + j * 16 + e;
                bf[j] = ld2(&Bt[col][kc + h * 8], &Bt[col][kc + 16 + h * 8]);
            }
            #pragma unroll
            for (int i = 0; i < 2; ++i)
                #pragma unroll
                for (int j = 0; j < 4; ++j)
                    acc[i][j] = wmma_bf16(af[i], bf[j], acc[i][j]);
        }
        __syncthreads();
    }

    #pragma unroll
    for (int i = 0; i < 2; ++i)
        #pragma unroll
        for (int j = 0; j < 4; ++j) {
            int col = n0 + wn * 64 + j * 16 + e;
            float bv = bias[col];
            #pragma unroll
            for (int v = 0; v < 8; ++v) {
                int row = m0 + wm * 32 + i * 16 + h * 8 + v;
                qkv[(size_t)row * C3 + col] = f2bf(acc[i][j][v] + bv);
            }
        }
}

// ===========================================================================
// Kernel 2: S = mask( (q @ k^T) / sqrt(C) )  (M=N=4096, K=1024), bf16 out
// ===========================================================================
__global__ __launch_bounds__(256)
void scores_kernel(const unsigned short* __restrict__ qkv,
                   const int* __restrict__ n_padd_p,
                   unsigned short* __restrict__ S) {
    if (blockIdx.x > blockIdx.y) return;   // strictly above diagonal: never read

    __shared__ __align__(16) unsigned short Qs[BM][LDST];   // [m][k]
    __shared__ __align__(16) unsigned short Ks[BM][LDST];   // [key n][k]

    const int tid  = threadIdx.x;
    const int lane = tid & 31;
    const int h    = lane >> 4;
    const int e    = lane & 15;
    const int wid  = tid >> 5;
    const int wm   = wid & 3;
    const int wn   = wid >> 2;
    const int m0   = blockIdx.y * BM;
    const int n0   = blockIdx.x * BN;
    const int np   = *n_padd_p;

    v8f acc[2][4] = {};

    for (int k0 = 0; k0 < C_DIM; k0 += BK) {
        // Q tile + K tile: straight bf16 copies, 1024 chunks each, 4/thread.
        #pragma unroll
        for (int i = 0; i < 4; ++i) {
            int p   = tid + i * 256;
            int row = p >> 3;            // 8 x 16B chunks per row
            int c8  = (p & 7) << 3;
            const unsigned short* src = qkv + (size_t)(m0 + row) * C3 + k0 + c8;
            if (i == 0 && k0 + BK < C_DIM) __builtin_prefetch(src + BK, 0, 1);
            cp16_g2l(src, &Qs[row][c8]);
        }
        #pragma unroll
        for (int i = 0; i < 4; ++i) {
            int p   = tid + i * 256;
            int row = p >> 3;
            int c8  = (p & 7) << 3;
            cp16_g2l(qkv + (size_t)(n0 + row) * C3 + C_DIM + k0 + c8, &Ks[row][c8]);
        }
        wait_async();
        __syncthreads();

        #pragma unroll
        for (int kc = 0; kc < BK; kc += 32) {
            v16us af[2], bf[4];
            #pragma unroll
            for (int i = 0; i < 2; ++i) {
                int row = wm * 32 + i * 16 + e;
                af[i] = ld2(&Qs[row][kc + h * 8], &Qs[row][kc + 16 + h * 8]);
            }
            #pragma unroll
            for (int j = 0; j < 4; ++j) {
                int col = wn * 64 + j * 16 + e;   // key index in tile
                bf[j] = ld2(&Ks[col][kc + h * 8], &Ks[col][kc + 16 + h * 8]);
            }
            #pragma unroll
            for (int i = 0; i < 2; ++i)
                #pragma unroll
                for (int j = 0; j < 4; ++j)
                    acc[i][j] = wmma_bf16(af[i], bf[j], acc[i][j]);
        }
        __syncthreads();
    }

    const float scale = 0.03125f;  // 1/sqrt(1024)
    #pragma unroll
    for (int i = 0; i < 2; ++i)
        #pragma unroll
        for (int j = 0; j < 4; ++j) {
            int col = n0 + wn * 64 + j * 16 + e;
            #pragma unroll
            for (int v = 0; v < 8; ++v) {
                int row = m0 + wm * 32 + i * 16 + h * 8 + v;
                bool keep = (col <= row) && (col >= np) && (row >= np);
                float val = keep ? acc[i][j][v] * scale : NEG_INF_F;
                S[(size_t)row * T_DIM + col] = f2bf(val);
            }
        }
}

// ===========================================================================
// Kernel 3: row softmax over causal prefix; att[r, c>r] = 0
// ===========================================================================
__global__ __launch_bounds__(256)
void softmax_kernel(const unsigned short* __restrict__ S,
                    unsigned short* __restrict__ att) {
    __shared__ float red[256];
    const int r   = blockIdx.x;
    const int tid = threadIdx.x;
    const unsigned short* srow = S + (size_t)r * T_DIM;
    unsigned short* arow       = att + (size_t)r * T_DIM;
    const int nvalid = r + 1;

    float m = -3.0e38f;
    for (int c = tid; c < nvalid; c += 256) m = fmaxf(m, bf2f(srow[c]));
    red[tid] = m; __syncthreads();
    for (int s = 128; s > 0; s >>= 1) {
        if (tid < s) red[tid] = fmaxf(red[tid], red[tid + s]);
        __syncthreads();
    }
    const float rmax = red[0];
    __syncthreads();

    float sum = 0.0f;
    for (int c = tid; c < nvalid; c += 256) sum += __expf(bf2f(srow[c]) - rmax);
    red[tid] = sum; __syncthreads();
    for (int s = 128; s > 0; s >>= 1) {
        if (tid < s) red[tid] += red[tid + s];
        __syncthreads();
    }
    const float inv = 1.0f / red[0];

    for (int c = tid; c < T_DIM; c += 256) {
        float p = (c < nvalid) ? __expf(bf2f(srow[c]) - rmax) * inv : 0.0f;
        arow[c] = f2bf(p);
    }
}

// ===========================================================================
// Kernel 4: y = att @ v   (M=4096, N=1024, K=4096), fp32 out; v = qkv[:,2C:3C]
// ===========================================================================
__global__ __launch_bounds__(256)
void av_gemm_kernel(const unsigned short* __restrict__ att,
                    const unsigned short* __restrict__ qkv,
                    float* __restrict__ y) {
    __shared__ __align__(16) unsigned short As[BM][LDST];   // [m][k]
    __shared__ __align__(16) unsigned short Bt[BN][LDST];   // [n][k] (transposed)

    const int tid  = threadIdx.x;
    const int lane = tid & 31;
    const int h    = lane >> 4;
    const int e    = lane & 15;
    const int wid  = tid >> 5;
    const int wm   = wid & 3;
    const int wn   = wid >> 2;
    const int m0   = blockIdx.y * BM;
    const int n0   = blockIdx.x * BN;

    v8f acc[2][4] = {};

    for (int k0 = 0; k0 < T_DIM; k0 += BK) {
        // A tile: att rows m0..+127, cols k0..+63 — straight copy, 4/thread.
        #pragma unroll
        for (int i = 0; i < 4; ++i) {
            int p   = tid + i * 256;
            int row = p >> 3;
            int c8  = (p & 7) << 3;
            const unsigned short* src = att + (size_t)(m0 + row) * T_DIM + k0 + c8;
            if (i == 0 && k0 + BK < T_DIM) __builtin_prefetch(src + BK, 0, 1);
            cp16_g2l(src, &As[row][c8]);
        }
        // B tile: v rows k0..+63, cols n0..+127, stored transposed. 4/thread.
        #pragma unroll
        for (int i = 0; i < 4; ++i) {
            int p   = tid + i * 256;
            int kr  = p >> 4;            // 16 chunks per k-row
            int c8  = (p & 15) << 3;
            const unsigned short* src =
                qkv + (size_t)(k0 + kr) * C3 + 2 * C_DIM + n0 + c8;
            if (i == 0 && k0 + BK < T_DIM) __builtin_prefetch(src + (size_t)BK * C3, 0, 1);
            union { uint4 q; unsigned short s[8]; } u;
            u.q = *reinterpret_cast<const uint4*>(src);
            #pragma unroll
            for (int ci = 0; ci < 8; ++ci) Bt[c8 + ci][kr] = u.s[ci];
        }
        wait_async();
        __syncthreads();

        #pragma unroll
        for (int kc = 0; kc < BK; kc += 32) {
            v16us af[2], bf[4];
            #pragma unroll
            for (int i = 0; i < 2; ++i) {
                int row = wm * 32 + i * 16 + e;
                af[i] = ld2(&As[row][kc + h * 8], &As[row][kc + 16 + h * 8]);
            }
            #pragma unroll
            for (int j = 0; j < 4; ++j) {
                int col = wn * 64 + j * 16 + e;
                bf[j] = ld2(&Bt[col][kc + h * 8], &Bt[col][kc + 16 + h * 8]);
            }
            #pragma unroll
            for (int i = 0; i < 2; ++i)
                #pragma unroll
                for (int j = 0; j < 4; ++j)
                    acc[i][j] = wmma_bf16(af[i], bf[j], acc[i][j]);
        }
        __syncthreads();
    }

    #pragma unroll
    for (int i = 0; i < 2; ++i)
        #pragma unroll
        for (int j = 0; j < 4; ++j) {
            int col = n0 + wn * 64 + j * 16 + e;
            #pragma unroll
            for (int v = 0; v < 8; ++v) {
                int row = m0 + wm * 32 + i * 16 + h * 8 + v;
                y[(size_t)row * C_DIM + col] = acc[i][j][v];
            }
        }
}

// ===========================================================================
extern "C" void kernel_launch(void* const* d_in, const int* in_sizes, int n_in,
                              void* d_out, int out_size, void* d_ws, size_t ws_size,
                              hipStream_t stream) {
    const float* x     = (const float*)d_in[0];   // [4096,1024]
    const float* W     = (const float*)d_in[1];   // [1024,3072]
    const float* bias  = (const float*)d_in[2];   // [3072]
    const int*   npadd = (const int*)d_in[3];     // scalar
    float*       y     = (float*)d_out;           // [4096,1024]

    unsigned short* qkv = (unsigned short*)d_ws;            // 24 MB
    unsigned short* S   = qkv + (size_t)T_DIM * C3;         // 32 MB
    unsigned short* att = S   + (size_t)T_DIM * T_DIM;      // 32 MB

    dim3 blk(256);
    qkv_gemm_kernel<<<dim3(C3 / BN,   T_DIM / BM), blk, 0, stream>>>(x, W, bias, qkv);
    scores_kernel  <<<dim3(T_DIM / BN, T_DIM / BM), blk, 0, stream>>>(qkv, npadd, S);
    softmax_kernel <<<dim3(T_DIM),                 blk, 0, stream>>>(S, att);
    av_gemm_kernel <<<dim3(C_DIM / BN, T_DIM / BM), blk, 0, stream>>>(att, qkv, y);
}
